// CML_Bi_Disen8_72722386256044
// MI455X (gfx1250) — matrix operand order repaired and checked by reading, not compiled, not run on hardware
//
#include <hip/hip_runtime.h>
#include <hip/hip_bf16.h>

// ---------------------------------------------------------------------------
// CML_Bi_Disen8 for MI455X (gfx1250): bf16 WMMA GEMMs + fused epilogues.
// B=4096, D=128, LU=10, LI=20.
// ---------------------------------------------------------------------------

#define BB 4096
#define DD 128
#define LUU 10
#define LII 20
#define XN (LII * DD) /* 2560 */

typedef __attribute__((ext_vector_type(16))) __bf16 v16bf;
typedef __attribute__((ext_vector_type(8)))  __bf16 v8bf;
typedef __attribute__((ext_vector_type(8)))  float  v8f;

// ---------------------------------------------------------------------------
// Zero scalar accumulators: acc[0]=mse, acc[1]=kld, acc[2]=mmi, acc[3]=zdiff
// ---------------------------------------------------------------------------
__global__ void init_acc_kernel(float* acc) {
    if (threadIdx.x < 4) acc[threadIdx.x] = 0.0f;
}

// ---------------------------------------------------------------------------
// Branch pooling: gather e + fe, masked softmax attention vs pop/int vectors,
// emit pooled operands as bf16 (row-major [B,128]) for the WMMA GEMMs.
// One block (128 threads) per batch row.
// ---------------------------------------------------------------------------
__global__ void pool_branch_kernel(const int* __restrict__ idx,
                                   const int* __restrict__ feat, int L,
                                   const float* __restrict__ emb,
                                   const float* __restrict__ ftab,
                                   const float* __restrict__ popv,
                                   const float* __restrict__ intv,
                                   __bf16* __restrict__ out_hot,
                                   __bf16* __restrict__ out_cold,
                                   float* __restrict__ pos_e) {
    __shared__ float fuse[(LII + 1) * DD];
    __shared__ float svp[DD], svi[DD];
    __shared__ float sch[LII + 1], scc[LII + 1];
    __shared__ int   smask[LII + 1];

    const int b = blockIdx.x;
    const int d = threadIdx.x;

    svp[d] = popv[d];
    svi[d] = intv[d];

    const int ei = idx[b];
    float ev = emb[(long long)ei * DD + d];
    fuse[d] = ev;
    if (pos_e) pos_e[(long long)b * DD + d] = ev;
    if (d == 0) smask[0] = 1;
    for (int l = 1; l <= L; ++l) {
        int f = feat[(long long)b * L + (l - 1)];
        fuse[l * DD + d] = ftab[(long long)f * DD + d];
        if (d == 0) smask[l] = f;
    }
    __syncthreads();

    if (d <= L) {
        float sh = 0.f, sc = 0.f;
        const float* row = &fuse[d * DD];
        for (int k = 0; k < DD; ++k) {
            float v = row[k];
            sh += v * svp[k];
            sc += v * svi[k];
        }
        sch[d] = sh;
        scc[d] = sc;
    }
    __syncthreads();

    // redundant per-thread softmax over <=21 scores (masked)
    float mh = -3.0e38f, mc = -3.0e38f;
    for (int l = 0; l <= L; ++l) {
        if (smask[l] != 0) {
            mh = fmaxf(mh, sch[l]);
            mc = fmaxf(mc, scc[l]);
        }
    }
    float dh = 0.f, dc = 0.f;
    for (int l = 0; l <= L; ++l) {
        if (smask[l] != 0) {
            dh += __expf(sch[l] - mh);
            dc += __expf(scc[l] - mc);
        }
    }
    float ph = 0.f, pc = 0.f;
    for (int l = 0; l <= L; ++l) {
        if (smask[l] != 0) {
            float v = fuse[l * DD + d];
            ph += (__expf(sch[l] - mh) / dh) * v;
            pc += (__expf(scc[l] - mc) / dc) * v;
        }
    }
    out_hot[(long long)b * DD + d]  = (__bf16)ph;
    out_cold[(long long)b * DD + d] = (__bf16)pc;
}

// ---------------------------------------------------------------------------
// Gather x = pos_fe reshaped [B, 2560] as bf16 (A operand of the enc GEMM).
// ---------------------------------------------------------------------------
__global__ void gather_x_kernel(const int* __restrict__ feat,
                                const float* __restrict__ ftab,
                                __bf16* __restrict__ xb) {
    const int b = blockIdx.x;
    for (int t = threadIdx.x; t < XN; t += blockDim.x) {
        int l = t >> 7, d = t & 127;
        int f = feat[(long long)b * LII + l];
        xb[(long long)b * XN + t] = (__bf16)ftab[(long long)f * DD + d];
    }
}

// ---------------------------------------------------------------------------
// Convert fp32 weight [K,N] row-major into bf16 B fragments in the CDNA5 WMMA
// 32x16 layout: per (kt,nt) tile, lane holds 16 K values for its column
// (lanes 0-15 -> K 0..7,16..23, lanes 16-31 -> K 8..15,24..31, N = lane&15).
// One wave per tile.
// ---------------------------------------------------------------------------
__global__ void conv_swz_kernel(const float* __restrict__ W,
                                __bf16* __restrict__ out, int K, int N) {
    const int ntn  = N >> 4;
    const int tile = blockIdx.x;
    const int kt   = tile / ntn;
    const int nt   = tile % ntn;
    const int lane = threadIdx.x;
    const int n    = nt * 16 + (lane & 15);
    const int kb   = kt * 32 + ((lane >> 4) ? 8 : 0);
    __bf16* dst = out + (((long long)kt * ntn + nt) * 32 + lane) * 16;
#pragma unroll
    for (int i = 0; i < 16; ++i)
        dst[i] = (__bf16)W[(long long)(kb + i + (i >= 8 ? 8 : 0)) * N + n];
}

// ---------------------------------------------------------------------------
// WMMA GEMM: C[M,N] = A[M,K](bf16,row-major) * Bswz(K,N) + bias, f32 accum.
//   mode 0: none   mode 1: ReLU   mode 2: accumulate sum((C - x_gather)^2)
// One wave computes one 16x16 tile via V_WMMA_F32_16X16X32_BF16.
// ---------------------------------------------------------------------------
__global__ void gemm_wmma_kernel(const __bf16* __restrict__ A,
                                 const __bf16* __restrict__ Bw,
                                 const float* __restrict__ bias,
                                 int M, int N, int K,
                                 float* __restrict__ Cf,
                                 __bf16* __restrict__ Cb,
                                 int mode,
                                 const int* __restrict__ feat,
                                 const float* __restrict__ ftab,
                                 float* __restrict__ acc) {
    const int ntn  = N >> 4;
    const int tile = blockIdx.x;
    const int mt   = tile / ntn;
    const int nt   = tile % ntn;
    const int lane = threadIdx.x;
    const int half = lane >> 4;
    const int l16  = lane & 15;

    const int arow = mt * 16 + l16;
    const __bf16* Arow = A + (long long)arow * K;
    const int kb0 = half ? 8 : 0;
    const int kt32 = K >> 5;

    v8f c = {};
    for (int kt = 0; kt < kt32; ++kt) {
        const int k0 = kt * 32 + kb0;
        v8bf a0 = *(const v8bf*)(Arow + k0);
        v8bf a1 = *(const v8bf*)(Arow + k0 + 16);
        v16bf a = __builtin_shufflevector(a0, a1, 0, 1, 2, 3, 4, 5, 6, 7,
                                          8, 9, 10, 11, 12, 13, 14, 15);
        const __bf16* bptr =
            Bw + (((long long)kt * ntn + nt) * 32 + lane) * 16;
        v16bf bf = *(const v16bf*)bptr;
        if (kt + 1 < kt32) {
            __builtin_prefetch(Arow + k0 + 32, 0, 1);
            __builtin_prefetch(bptr + (long long)ntn * 32 * 16, 0, 1);
        }
        c = __builtin_amdgcn_wmma_f32_16x16x32_bf16(
            false, a, false, bf, (short)0, c, false, false);
    }

    const int col = nt * 16 + l16;
    const float bv = bias ? bias[col] : 0.f;
    float part = 0.f;
#pragma unroll
    for (int r = 0; r < 8; ++r) {
        const int row = mt * 16 + r + half * 8;
        float v = c[r] + bv;
        if (mode == 1) v = fmaxf(v, 0.f);
        if (Cf) Cf[(long long)row * N + col] = v;
        if (Cb) Cb[(long long)row * N + col] = (__bf16)v;
        if (mode == 2) {
            int l = col >> 7, dd = col & 127;
            int f = feat[(long long)row * LII + l];
            float xv = ftab[(long long)f * DD + dd];
            float df = v - xv;
            part += df * df;
        }
    }
    if (mode == 2) {
        for (int off = 16; off; off >>= 1) part += __shfl_xor(part, off, 32);
        if (lane == 0) atomicAdd(acc, part);
    }
}

// ---------------------------------------------------------------------------
// LDS-staged WMMA GEMM for the large-K encoder GEMM.
// Block = 256 threads = 8 waves; block computes 128x16 of C (8 m-tiles that
// share one n-tile). Wave 0 double-buffers the 1KB B fragment tile per K-step
// into LDS with GLOBAL_LOAD_ASYNC_TO_LDS_B128 (ASYNCcnt path), all 8 waves
// consume it via ds_load_b128 -> B traffic cut 8x and the async copies ride a
// counter independent of the A-operand LOADcnt.
// ---------------------------------------------------------------------------
__global__ void gemm_wmma_lds_kernel(const __bf16* __restrict__ A,
                                     const __bf16* __restrict__ Bw,
                                     const float* __restrict__ bias,
                                     int M, int N, int K,
                                     float* __restrict__ Cf,
                                     __bf16* __restrict__ Cb,
                                     int relu) {
    __shared__ __align__(32) __bf16 sB[2][32 * 16];

    const int ntn  = N >> 4;
    const int nt   = blockIdx.x % ntn;
    const int msb  = blockIdx.x / ntn;       // super-block of 8 m-tiles
    const int wave = threadIdx.x >> 5;
    const int lane = threadIdx.x & 31;
    const int half = lane >> 4;
    const int l16  = lane & 15;

    const int mt   = msb * 8 + wave;
    const int arow = mt * 16 + l16;
    const __bf16* Arow = A + (long long)arow * K;
    const int kb0  = half ? 8 : 0;
    const int kt32 = K >> 5;

    // stage B fragment tile for K-step kt into LDS buffer buf (wave 0 only)
    auto stage_b = [&](int buf, int kt) {
        const __bf16* src =
            Bw + (((long long)kt * ntn + nt) * 32 + lane) * 16;
        unsigned dst = (unsigned)(size_t)(&sB[buf][lane * 16]);
        unsigned long long g0 = (unsigned long long)(size_t)src;
        asm volatile("global_load_async_to_lds_b128 %0, %1, off"
                     :: "v"(dst), "v"(g0) : "memory");
        asm volatile("global_load_async_to_lds_b128 %0, %1, off"
                     :: "v"(dst + 16u), "v"(g0 + 16ull) : "memory");
    };

    if (wave == 0) stage_b(0, 0);

    v8f c = {};
    for (int kt = 0; kt < kt32; ++kt) {
        const int cur = kt & 1;
        if (wave == 0)
            asm volatile("s_wait_asynccnt 0x0" ::: "memory");
        __syncthreads();                 // buf[cur] ready for all waves
        if (wave == 0 && (kt + 1) < kt32)
            stage_b(cur ^ 1, kt + 1);    // overlap next B copy with math

        const int k0 = kt * 32 + kb0;
        v8bf a0 = *(const v8bf*)(Arow + k0);
        v8bf a1 = *(const v8bf*)(Arow + k0 + 16);
        v16bf a = __builtin_shufflevector(a0, a1, 0, 1, 2, 3, 4, 5, 6, 7,
                                          8, 9, 10, 11, 12, 13, 14, 15);
        v16bf bf = *(const v16bf*)(&sB[cur][lane * 16]);
        c = __builtin_amdgcn_wmma_f32_16x16x32_bf16(
            false, a, false, bf, (short)0, c, false, false);
        __syncthreads();                 // all done with buf[cur] before reuse
    }

    const int col = nt * 16 + l16;
    const float bv = bias ? bias[col] : 0.f;
#pragma unroll
    for (int r = 0; r < 8; ++r) {
        const int row = mt * 16 + r + half * 8;
        float v = c[r] + bv;
        if (relu) v = fmaxf(v, 0.f);
        if (Cf) Cf[(long long)row * N + col] = v;
        if (Cb) Cb[(long long)row * N + col] = (__bf16)v;
    }
}

// ---------------------------------------------------------------------------
// z = eps*exp(0.5*logvar) + mu ; accumulate KLD. One block (128) per row.
// ---------------------------------------------------------------------------
__global__ void z_kld_kernel(const float* __restrict__ mu,
                             const float* __restrict__ lv,
                             const float* __restrict__ eps,
                             float* __restrict__ z, __bf16* __restrict__ zb,
                             float* __restrict__ acc) {
    __shared__ float red[DD];
    const int b = blockIdx.x, d = threadIdx.x;
    const long long i = (long long)b * DD + d;
    const float m = mu[i], l = lv[i], e = eps[i];
    const float zz = e * __expf(0.5f * l) + m;
    z[i]  = zz;
    zb[i] = (__bf16)zz;
    red[d] = -0.5f * (1.0f + l - m * m - __expf(l));
    __syncthreads();
    for (int s = 64; s; s >>= 1) {
        if (d < s) red[d] += red[d + s];
        __syncthreads();
    }
    if (d == 0) atomicAdd(acc, red[0]);
}

// ---------------------------------------------------------------------------
// MMI + ||z - pos_e||^2 accumulation. One block (128) per row.
// ---------------------------------------------------------------------------
__global__ void mmi_zd_kernel(const float* __restrict__ z,
                              const float* __restrict__ pe,
                              const float* __restrict__ emb_item,
                              const int* __restrict__ comp,
                              float* __restrict__ mmi_acc,
                              float* __restrict__ zd_acc) {
    __shared__ float r1[DD], r2[DD], r3[DD];
    const int b = blockIdx.x, d = threadIdx.x;
    const long long i = (long long)b * DD + d;
    const float zv = z[i], pv = pe[i];
    const float ce = emb_item[(long long)comp[b] * DD + d];
    r1[d] = zv * pv;
    r2[d] = zv * ce;
    const float df = zv - pv;
    r3[d] = df * df;
    __syncthreads();
    for (int s = 64; s; s >>= 1) {
        if (d < s) {
            r1[d] += r1[d + s];
            r2[d] += r2[d + s];
            r3[d] += r3[d + s];
        }
        __syncthreads();
    }
    if (d == 0) {
        float x = r1[0] - r2[0];
        // stable log_sigmoid(x) = min(x,0) - log1p(exp(-|x|))
        float ls = fminf(x, 0.f) - log1pf(__expf(-fabsf(x)));
        atomicAdd(mmi_acc, -ls);
        atomicAdd(zd_acc, r3[0]);
    }
}

// ---------------------------------------------------------------------------
// Triplet losses: relu(||u-p||^2 - ||u-n||^2 + 1) for hot and cold.
// ---------------------------------------------------------------------------
__global__ void losses_kernel(const float* __restrict__ uh,
                              const float* __restrict__ ph,
                              const float* __restrict__ nh,
                              const float* __restrict__ uc,
                              const float* __restrict__ pc,
                              const float* __restrict__ nc,
                              float* __restrict__ out) {
    __shared__ float r1[DD], r2[DD], r3[DD], r4[DD];
    const int b = blockIdx.x, d = threadIdx.x;
    const long long i = (long long)b * DD + d;
    float a, bb2;
    a = uh[i] - ph[i]; r1[d] = a * a;
    a = uh[i] - nh[i]; r2[d] = a * a;
    bb2 = uc[i] - pc[i]; r3[d] = bb2 * bb2;
    bb2 = uc[i] - nc[i]; r4[d] = bb2 * bb2;
    __syncthreads();
    for (int s = 64; s; s >>= 1) {
        if (d < s) {
            r1[d] += r1[d + s];
            r2[d] += r2[d + s];
            r3[d] += r3[d + s];
            r4[d] += r4[d + s];
        }
        __syncthreads();
    }
    if (d == 0) {
        out[b]      = fmaxf(r1[0] - r2[0] + 1.0f, 0.f);
        out[BB + b] = fmaxf(r3[0] - r4[0] + 1.0f, 0.f);
    }
}

__global__ void finalize_kernel(const float* __restrict__ acc,
                                float* __restrict__ out) {
    // item_vae_loss = mmi + mse/LI + kld + zdiff
    out[2 * BB] = acc[2] + acc[0] / (float)LII + acc[1] + acc[3];
}

// ---------------------------------------------------------------------------
// Host launcher
// ---------------------------------------------------------------------------
extern "C" void kernel_launch(void* const* d_in, const int* in_sizes, int n_in,
                              void* d_out, int out_size, void* d_ws,
                              size_t ws_size, hipStream_t stream) {
    (void)in_sizes; (void)n_in; (void)out_size; (void)ws_size;

    const int*   user_idx   = (const int*)d_in[0];
    const int*   user_feat  = (const int*)d_in[1];
    const int*   pos_idx    = (const int*)d_in[2];
    const int*   pos_feat   = (const int*)d_in[3];
    const int*   neg_idx    = (const int*)d_in[4];   // (B,1) -> B
    const int*   neg_feat   = (const int*)d_in[5];   // (B,1,LI) -> (B,LI)
    const int*   comp_idx   = (const int*)d_in[6];
    const float* eps        = (const float*)d_in[7];
    const float* emb_user   = (const float*)d_in[8];
    const float* emb_item   = (const float*)d_in[9];
    const float* fe_user    = (const float*)d_in[10];
    const float* fe_item    = (const float*)d_in[11];
    const float* pop_u      = (const float*)d_in[12];
    const float* int_u      = (const float*)d_in[13];
    const float* pop_i      = (const float*)d_in[14];
    const float* int_i      = (const float*)d_in[15];
    const float* W_pop      = (const float*)d_in[16];
    const float* b_pop      = (const float*)d_in[17];
    const float* W_int      = (const float*)d_in[18];
    const float* b_int      = (const float*)d_in[19];
    const float* enc_W      = (const float*)d_in[20];
    const float* enc_b      = (const float*)d_in[21];
    const float* dec_W      = (const float*)d_in[22];
    const float* dec_b      = (const float*)d_in[23];
    const float* fc11_W     = (const float*)d_in[24];
    const float* fc11_b     = (const float*)d_in[25];
    const float* fc12_W     = (const float*)d_in[26];
    const float* fc12_b     = (const float*)d_in[27];
    float* out = (float*)d_out;

    // ---- workspace carve-up (256B aligned) ----
    char* wp = (char*)d_ws;
    auto carve = [&](size_t bytes) -> void* {
        void* p = (void*)wp;
        wp += (bytes + 255) & ~(size_t)255;
        return p;
    };
    __bf16* x_bf     = (__bf16*)carve((size_t)BB * XN * 2);
    __bf16* encW_bf  = (__bf16*)carve((size_t)XN * DD * 2);
    __bf16* decW_bf  = (__bf16*)carve((size_t)DD * XN * 2);
    __bf16* fc11_bf  = (__bf16*)carve((size_t)DD * DD * 2);
    __bf16* fc12_bf  = (__bf16*)carve((size_t)DD * DD * 2);
    __bf16* Wpop_bf  = (__bf16*)carve((size_t)DD * DD * 2);
    __bf16* Wint_bf  = (__bf16*)carve((size_t)DD * DD * 2);
    __bf16* pl_uh    = (__bf16*)carve((size_t)BB * DD * 2);
    __bf16* pl_uc    = (__bf16*)carve((size_t)BB * DD * 2);
    __bf16* pl_ph    = (__bf16*)carve((size_t)BB * DD * 2);
    __bf16* pl_pc    = (__bf16*)carve((size_t)BB * DD * 2);
    __bf16* pl_nh    = (__bf16*)carve((size_t)BB * DD * 2);
    __bf16* pl_nc    = (__bf16*)carve((size_t)BB * DD * 2);
    float*  u_hot    = (float*)carve((size_t)BB * DD * 4);
    float*  u_cold   = (float*)carve((size_t)BB * DD * 4);
    float*  p_hot    = (float*)carve((size_t)BB * DD * 4);
    float*  p_cold   = (float*)carve((size_t)BB * DD * 4);
    float*  n_hot    = (float*)carve((size_t)BB * DD * 4);
    float*  n_cold   = (float*)carve((size_t)BB * DD * 4);
    __bf16* h1_bf    = (__bf16*)carve((size_t)BB * DD * 2);
    float*  mu       = (float*)carve((size_t)BB * DD * 4);
    float*  logvar   = (float*)carve((size_t)BB * DD * 4);
    float*  z        = (float*)carve((size_t)BB * DD * 4);
    __bf16* z_bf     = (__bf16*)carve((size_t)BB * DD * 2);
    float*  pos_e    = (float*)carve((size_t)BB * DD * 4);
    float*  acc      = (float*)carve(4 * sizeof(float));

    init_acc_kernel<<<1, 32, 0, stream>>>(acc);

    // ---- attention pooling (bf16 pooled operands) ----
    pool_branch_kernel<<<BB, DD, 0, stream>>>(user_idx, user_feat, LUU,
        emb_user, fe_user, pop_u, int_u, pl_uh, pl_uc, nullptr);
    pool_branch_kernel<<<BB, DD, 0, stream>>>(pos_idx, pos_feat, LII,
        emb_item, fe_item, pop_i, int_i, pl_ph, pl_pc, pos_e);
    pool_branch_kernel<<<BB, DD, 0, stream>>>(neg_idx, neg_feat, LII,
        emb_item, fe_item, pop_i, int_i, pl_nh, pl_nc, nullptr);

    gather_x_kernel<<<BB, 256, 0, stream>>>(pos_feat, fe_item, x_bf);

    // ---- weight conversion into WMMA B-fragment layout ----
    conv_swz_kernel<<<(DD / 32) * (DD / 16), 32, 0, stream>>>(W_pop, Wpop_bf, DD, DD);
    conv_swz_kernel<<<(DD / 32) * (DD / 16), 32, 0, stream>>>(W_int, Wint_bf, DD, DD);
    conv_swz_kernel<<<(XN / 32) * (DD / 16), 32, 0, stream>>>(enc_W, encW_bf, XN, DD);
    conv_swz_kernel<<<(DD / 32) * (XN / 16), 32, 0, stream>>>(dec_W, decW_bf, DD, XN);
    conv_swz_kernel<<<(DD / 32) * (DD / 16), 32, 0, stream>>>(fc11_W, fc11_bf, DD, DD);
    conv_swz_kernel<<<(DD / 32) * (DD / 16), 32, 0, stream>>>(fc12_W, fc12_bf, DD, DD);

    const int tiles_BD = (BB / 16) * (DD / 16);   // 2048
    const int tiles_BX = (BB / 16) * (XN / 16);   // 40960

    // ---- six pooled @ W GEMMs (M=4096, K=128, N=128) ----
    gemm_wmma_kernel<<<tiles_BD, 32, 0, stream>>>(pl_uh, Wpop_bf, b_pop,
        BB, DD, DD, u_hot, nullptr, 0, nullptr, nullptr, nullptr);
    gemm_wmma_kernel<<<tiles_BD, 32, 0, stream>>>(pl_uc, Wint_bf, b_int,
        BB, DD, DD, u_cold, nullptr, 0, nullptr, nullptr, nullptr);
    gemm_wmma_kernel<<<tiles_BD, 32, 0, stream>>>(pl_ph, Wpop_bf, b_pop,
        BB, DD, DD, p_hot, nullptr, 0, nullptr, nullptr, nullptr);
    gemm_wmma_kernel<<<tiles_BD, 32, 0, stream>>>(pl_pc, Wint_bf, b_int,
        BB, DD, DD, p_cold, nullptr, 0, nullptr, nullptr, nullptr);
    gemm_wmma_kernel<<<tiles_BD, 32, 0, stream>>>(pl_nh, Wpop_bf, b_pop,
        BB, DD, DD, n_hot, nullptr, 0, nullptr, nullptr, nullptr);
    gemm_wmma_kernel<<<tiles_BD, 32, 0, stream>>>(pl_nc, Wint_bf, b_int,
        BB, DD, DD, n_cold, nullptr, 0, nullptr, nullptr, nullptr);

    // ---- VAE encoder: h1 = relu(x @ enc_W + enc_b)  (K=2560) ----
    // LDS-staged variant: 8 waves share the B tile via async-to-LDS copies.
    gemm_wmma_lds_kernel<<<(BB / 128) * (DD / 16), 256, 0, stream>>>(
        x_bf, encW_bf, enc_b, BB, DD, XN, nullptr, h1_bf, 1);
    // mu / logvar
    gemm_wmma_kernel<<<tiles_BD, 32, 0, stream>>>(h1_bf, fc11_bf, fc11_b,
        BB, DD, DD, mu, nullptr, 0, nullptr, nullptr, nullptr);
    gemm_wmma_kernel<<<tiles_BD, 32, 0, stream>>>(h1_bf, fc12_bf, fc12_b,
        BB, DD, DD, logvar, nullptr, 0, nullptr, nullptr, nullptr);

    // z + KLD
    z_kld_kernel<<<BB, DD, 0, stream>>>(mu, logvar, eps, z, z_bf, &acc[1]);

    // ---- decoder GEMM fused with sum((recon - x)^2): recon never stored ----
    gemm_wmma_kernel<<<tiles_BX, 32, 0, stream>>>(z_bf, decW_bf, dec_b,
        BB, XN, DD, nullptr, nullptr, 2, pos_feat, fe_item, &acc[0]);

    // MMI + ||z - pos_e||^2
    mmi_zd_kernel<<<BB, DD, 0, stream>>>(z, pos_e, emb_item, comp_idx,
                                         &acc[2], &acc[3]);

    // triplet losses -> out[0:4096], out[4096:8192]
    losses_kernel<<<BB, DD, 0, stream>>>(u_hot, p_hot, n_hot,
                                         u_cold, p_cold, n_cold, out);

    finalize_kernel<<<1, 1, 0, stream>>>(acc, out);
}
